// Attention_17093969838576
// MI455X (gfx1250) — compile-verified, matrix operand order checked by reading, not run on hardware
//
#include <hip/hip_runtime.h>

typedef __attribute__((ext_vector_type(16))) __bf16 v16bf;
typedef __attribute__((ext_vector_type(8)))  __bf16 v8bf;
typedef __attribute__((ext_vector_type(8)))  float  v8f;

#define B_   4
#define S_   1024
#define HQ_  32
#define HK_  8
#define D_   128
#define BS_  256
#define NBT  (S_ / BS_)
#define WAVES 4
// 1/sqrt(128) * log2(e): softmax computed in base-2 (v_exp_f32 is native exp2)
#define QSCALE (0.08838834764831845f * 1.4426950408889634f)

__device__ __forceinline__ unsigned pack2(float lo, float hi) {
    union { float f; unsigned u; } a, b; a.f = lo; b.f = hi;
    return __builtin_amdgcn_perm(b.u, a.u, 0x07060302u);   // {bf16(hi), bf16(lo)}
}
__device__ __forceinline__ __bf16 f2bf(float f) {          // truncation: hi16 of f32
    union { float f; unsigned u; } a; a.f = f;
    union { unsigned short s; __bf16 b; } c; c.s = (unsigned short)(a.u >> 16);
    return c.b;
}
template <int XM>
__device__ __forceinline__ float row_xor_f(float x) {      // DPP row_xmask butterfly
    union { float f; int i; } a, r; a.f = x;
    r.i = __builtin_amdgcn_mov_dpp(a.i, 0x160 + XM, 0xF, 0xF, true);
    return r.f;
}
__device__ __forceinline__ v16bf load_frag_row_scaled(const float* __restrict__ rowp,
                                                      int hh, float sc) {
    const float4* p0 = (const float4*)(rowp + hh * 8);
    const float4* p1 = (const float4*)(rowp + 16 + hh * 8);
    float4 a0 = p0[0], a1 = p0[1], b0 = p1[0], b1 = p1[1];
    union { v16bf v; unsigned u[8]; } r;
    r.u[0] = pack2(a0.x*sc, a0.y*sc); r.u[1] = pack2(a0.z*sc, a0.w*sc);
    r.u[2] = pack2(a1.x*sc, a1.y*sc); r.u[3] = pack2(a1.z*sc, a1.w*sc);
    r.u[4] = pack2(b0.x*sc, b0.y*sc); r.u[5] = pack2(b0.z*sc, b0.w*sc);
    r.u[6] = pack2(b1.x*sc, b1.y*sc); r.u[7] = pack2(b1.z*sc, b1.w*sc);
    return r.v;
}
__device__ __forceinline__ v16bf cat8(v8bf lo, v8bf hi) {
    return __builtin_shufflevector(lo, hi, 0,1,2,3,4,5,6,7,8,9,10,11,12,13,14,15);
}

__global__ __launch_bounds__(32 * WAVES)
void fa_fwd_kernel(const float* __restrict__ q,
                   const float* __restrict__ kc,
                   const float* __restrict__ vc,
                   const int*   __restrict__ bt,
                   float* __restrict__ out)
{
    // single-buffered 64-key tiles (two barriers per iteration), 40 KB total
    __shared__ __align__(16) __bf16 lds_k[64][D_];        // 16 KB
    __shared__ __align__(16) __bf16 lds_v[D_][64];        // 16 KB (V^T)
    __shared__ __align__(16) __bf16 lds_p[WAVES][16][64]; //  8 KB (per-wave)

    const int tid  = threadIdx.x;
    const int lane = tid & 31;
    const int wave = tid >> 5;
    const int wg   = blockIdx.x;
    const int qtg  = wg & 15;
    const int h    = (wg >> 4) & 31;
    const int b    = wg >> 9;
    const int hk   = h >> 2;
    const int qr0  = (qtg * WAVES + wave) * 16;
    const int hh   = lane >> 4;
    const int lid  = lane & 15;

    // ---- Q tile: 4 bf16 A-fragments, pre-scaled by SCALE*log2e ----
    const float* qrow_ptr = q + ((size_t)(b * S_ + qr0 + lid) * HQ_ + h) * D_;
    v16bf qf[4];
    #pragma unroll
    for (int c = 0; c < 4; ++c) qf[c] = load_frag_row_scaled(qrow_ptr + c * 32, hh, QSCALE);

    union { v16bf v; unsigned u[8]; } onesu;
    #pragma unroll
    for (int i = 0; i < 8; ++i) onesu.u[i] = 0x3F803F80u;  // bf16 1.0 pair
    const v16bf ones_f = onesu.v;

    v8f o[8] = {};
    float m_r[8], l_r[8];
    #pragma unroll
    for (int r = 0; r < 8; ++r) { m_r[r] = -1e30f; l_r[r] = 0.0f; }

    const int* btb = bt + b * NBT;
    const int n_iters = qtg + 1;         // 64 keys per iteration, uniform per block

    const int skey = tid >> 1;           // K staging: key, 2 lanes each
    const int sd   = (tid & 1) * 64;     // K staging: 64-d half
    const int vkey = (tid & 31) * 2;     // V staging: adjacent key pair
    const int vd   = (tid >> 5) * 32;    // V staging: 32-d quarter

    for (int it = 0; it < n_iters; ++it) {
        const int kb0 = it * 64;         // 64-key block never crosses a 256-key page
        const int page = __builtin_amdgcn_readfirstlane(btb[kb0 >> 8]);
        const size_t krow0 = (size_t)page * BS_ + (kb0 & (BS_ - 1));

        // ---- cooperative K staging: 64 floats/lane ----
        {
            const float4* kp4 = (const float4*)(kc + ((krow0 + skey) * HK_ + hk) * D_ + sd);
            uint4* dst = (uint4*)&lds_k[skey][sd];
            #pragma unroll
            for (int i = 0; i < 8; ++i) {
                float4 fa = kp4[2 * i], fb = kp4[2 * i + 1];
                dst[i] = make_uint4(pack2(fa.x, fa.y), pack2(fa.z, fa.w),
                                    pack2(fb.x, fb.y), pack2(fb.z, fb.w));
            }
        }
        // ---- cooperative V^T staging: key pair packed per dword ----
        {
            const float* vpA = vc + ((krow0 + vkey) * HK_ + hk) * D_ + vd;
            const float* vpB = vpA + (size_t)HK_ * D_;
            #pragma unroll
            for (int j = 0; j < 8; ++j) {
                float4 a  = ((const float4*)vpA)[j];
                float4 bb = ((const float4*)vpB)[j];
                const int d0 = vd + j * 4;
                *(unsigned*)&lds_v[d0 + 0][vkey] = pack2(a.x, bb.x);
                *(unsigned*)&lds_v[d0 + 1][vkey] = pack2(a.y, bb.y);
                *(unsigned*)&lds_v[d0 + 2][vkey] = pack2(a.z, bb.z);
                *(unsigned*)&lds_v[d0 + 3][vkey] = pack2(a.w, bb.w);
            }
        }
        if (it + 1 < n_iters) {
            const int np = __builtin_amdgcn_readfirstlane(btb[(kb0 + 64) >> 8]);
            const float* pfp = kc + (((size_t)np * BS_ + ((kb0 + 64) & (BS_ - 1)) + skey) * HK_ + hk) * D_ + sd;
            __builtin_prefetch(pfp, 0, 1);
        }

        __syncthreads();

        // ---- S = Q*K^T: four 16-key tiles, 16 WMMAs ----
        v8f s[4];
        #pragma unroll
        for (int t4 = 0; t4 < 4; ++t4) {
            v16bf kf[4];
            #pragma unroll
            for (int c = 0; c < 4; ++c)
                kf[c] = cat8(*(const v8bf*)&lds_k[t4 * 16 + lid][c * 32 + hh * 8],
                             *(const v8bf*)&lds_k[t4 * 16 + lid][c * 32 + 16 + hh * 8]);
            v8f acc = {};
            #pragma unroll
            for (int c = 0; c < 4; ++c)
                acc = __builtin_amdgcn_wmma_f32_16x16x32_bf16(false, qf[c], false, kf[c],
                                                              (short)0, acc, false, false);
            s[t4] = acc;
        }

        // ---- causal mask only when block touches this wave's diagonal ----
        if (kb0 + 64 > qr0) {                      // wave-uniform branch
            #pragma unroll
            for (int t4 = 0; t4 < 4; ++t4)
                #pragma unroll
                for (int r = 0; r < 8; ++r)
                    if (kb0 + t4 * 16 + lid > qr0 + r + hh * 8) s[t4][r] = -1e30f;
        }

        // ---- rowmax via DPP butterfly ----
        float mx[8];
        #pragma unroll
        for (int r = 0; r < 8; ++r)
            mx[r] = fmaxf(fmaxf(s[0][r], s[1][r]), fmaxf(s[2][r], s[3][r]));
        #pragma unroll
        for (int r = 0; r < 8; ++r) mx[r] = fmaxf(mx[r], row_xor_f<8>(mx[r]));
        #pragma unroll
        for (int r = 0; r < 8; ++r) mx[r] = fmaxf(mx[r], row_xor_f<4>(mx[r]));
        #pragma unroll
        for (int r = 0; r < 8; ++r) mx[r] = fmaxf(mx[r], row_xor_f<2>(mx[r]));
        #pragma unroll
        for (int r = 0; r < 8; ++r) mx[r] = fmaxf(mx[r], row_xor_f<1>(mx[r]));

        // ---- rescale only when some row found a new max (wave-uniform skip) ----
        bool need = false;
        #pragma unroll
        for (int r = 0; r < 8; ++r) need = need || (mx[r] > m_r[r]);
        if (__ballot(need) != 0ull) {
            #pragma unroll
            for (int r = 0; r < 8; ++r) {
                float nm   = fmaxf(m_r[r], mx[r]);
                float corr = __builtin_amdgcn_exp2f(m_r[r] - nm);
                m_r[r] = nm;
                l_r[r] *= corr;
                #pragma unroll
                for (int t = 0; t < 8; ++t) o[t][r] *= corr;
            }
        }

        // ---- P = exp2(S - m), straight into per-wave LDS (restripe) ----
        #pragma unroll
        for (int t4 = 0; t4 < 4; ++t4)
            #pragma unroll
            for (int r = 0; r < 8; ++r)
                lds_p[wave][r + hh * 8][t4 * 16 + lid] =
                    f2bf(__builtin_amdgcn_exp2f(s[t4][r] - m_r[r]));
        asm volatile("s_wait_dscnt 0" ::: "memory");   // per-wave LDS, in-order

        v16bf pfA = cat8(*(const v8bf*)&lds_p[wave][lid][hh * 8],
                         *(const v8bf*)&lds_p[wave][lid][16 + hh * 8]);
        v16bf pfB = cat8(*(const v8bf*)&lds_p[wave][lid][32 + hh * 8],
                         *(const v8bf*)&lds_p[wave][lid][48 + hh * 8]);

        // rowsum(P) in two WMMAs (P x ones, accumulate)
        v8f zc = {};
        v8f rsum = __builtin_amdgcn_wmma_f32_16x16x32_bf16(false, pfA, false, ones_f,
                                                           (short)0, zc, false, false);
        rsum = __builtin_amdgcn_wmma_f32_16x16x32_bf16(false, pfB, false, ones_f,
                                                       (short)0, rsum, false, false);
        #pragma unroll
        for (int r = 0; r < 8; ++r) l_r[r] += rsum[r];

        // ---- O += P*V: 16 WMMAs over the 128-wide d dimension ----
        #pragma unroll
        for (int t = 0; t < 8; ++t) {
            const int dcol = t * 16 + lid;
            v16bf vfA = cat8(*(const v8bf*)&lds_v[dcol][hh * 8],
                             *(const v8bf*)&lds_v[dcol][16 + hh * 8]);
            v16bf vfB = cat8(*(const v8bf*)&lds_v[dcol][32 + hh * 8],
                             *(const v8bf*)&lds_v[dcol][48 + hh * 8]);
            o[t] = __builtin_amdgcn_wmma_f32_16x16x32_bf16(false, pfA, false, vfA,
                                                           (short)0, o[t], false, false);
            o[t] = __builtin_amdgcn_wmma_f32_16x16x32_bf16(false, pfB, false, vfB,
                                                           (short)0, o[t], false, false);
        }

        __syncthreads();   // all reads of lds_k/lds_v done before next staging
    }

    // ---- epilogue: normalize and store ----
    #pragma unroll
    for (int r = 0; r < 8; ++r) {
        const float inv = 1.0f / l_r[r];
        const int row = qr0 + r + hh * 8;
        float* orow = out + ((size_t)(b * S_ + row) * HQ_ + h) * D_;
        #pragma unroll
        for (int t = 0; t < 8; ++t)
            orow[t * 16 + lid] = o[t][r] * inv;
    }
}

extern "C" void kernel_launch(void* const* d_in, const int* in_sizes, int n_in,
                              void* d_out, int out_size, void* d_ws, size_t ws_size,
                              hipStream_t stream) {
    const float* q  = (const float*)d_in[0];
    const float* kc = (const float*)d_in[1];
    const float* vc = (const float*)d_in[2];
    const int*   bt = (const int*)d_in[3];
    dim3 grid(B_ * HQ_ * (S_ / 16 / WAVES));   // 2048 blocks of 128 threads
    fa_fwd_kernel<<<grid, 32 * WAVES, 0, stream>>>(q, kc, vc, bt, (float*)d_out);
}
